// INT_QuantModule_DiffAE_LoRA_31748398252838
// MI455X (gfx1250) — matrix operand order repaired and checked by reading, not compiled, not run on hardware
//
#include <hip/hip_runtime.h>

typedef __attribute__((ext_vector_type(16))) _Float16 v16h;
typedef __attribute__((ext_vector_type(8)))  _Float16 v8h;
typedef __attribute__((ext_vector_type(8)))  float    v8f;
typedef __attribute__((ext_vector_type(4)))  float    v4f;
typedef __attribute__((ext_vector_type(4)))  int      v4i;
typedef __attribute__((ext_vector_type(8)))  unsigned v8u;

#define OC_DIM    8192
#define IC_DIM    8192
#define B_DIM     64
#define PACK_ROWS 4096                      // packed weight rows (2 nibbles each)

// d_ws layout (bytes)
#define XH_OFF    0u                        // f16 x   [64][8192]  : 1 MB
#define RS_OFF    (1u << 20)                // rowsum  [64] f32
#define PART_OFF  ((1u << 20) + 4096u)      // partial S buffers, 2 MB each
#define PART_SZ   ((size_t)B_DIM * OC_DIM * 4)

// ---------------------------------------------------------------------------
// Prep: x (f32) -> xh (f16, row-major) once; rowsum[m] = sum_k f16(x[m,k]).
// ---------------------------------------------------------------------------
__global__ __launch_bounds__(256) void prep_x(const float* __restrict__ x,
                                              _Float16* __restrict__ xh,
                                              float* __restrict__ rowsum)
{
    __shared__ float sred[256];
    const int row = blockIdx.x;
    const int tid = threadIdx.x;
    const float* xr = x  + (size_t)row * IC_DIM;
    _Float16*    hr = xh + (size_t)row * IC_DIM;
    const int base = tid * 32;                      // 8192/256 = 32 per thread

    float s = 0.f;
#pragma unroll
    for (int i = 0; i < 32; i += 8) {
        v4f v0 = *(const v4f*)(xr + base + i);
        v4f v1 = *(const v4f*)(xr + base + i + 4);
        v8h h;
#pragma unroll
        for (int j = 0; j < 4; ++j) {
            h[j]     = (_Float16)v0[j];
            h[4 + j] = (_Float16)v1[j];
        }
#pragma unroll
        for (int j = 0; j < 8; ++j) s += (float)h[j];   // sum of f16-rounded x
        *(v8h*)(hr + base + i) = h;
    }
    sred[tid] = s;
    __syncthreads();
#pragma unroll
    for (int off = 128; off > 0; off >>= 1) {
        if (tid < off) sred[tid] += sred[tid + off];
        __syncthreads();
    }
    if (tid == 0) rowsum[row] = sred[0];
}

// ---------------------------------------------------------------------------
// Main: wave w exclusively owns 16 packed columns and all four M tiles
// (8 WMMAs / weight fetch). 2-deep register double buffering: each phase
// issues the NEXT tile's 12 loads, then converts + WMMAs the CURRENT tile.
// ---------------------------------------------------------------------------
__global__ __launch_bounds__(128) void w4_wmma_main(
    const _Float16* __restrict__ xh,   // [64][8192] f16
    const int*      __restrict__ wp,   // [4096][8192] int32 byte values
    float*          __restrict__ part, // ksplit x [64][8192] f32 partial S
    int kchunk)                        // IC_DIM / ksplit (multiple of 64)
{
    const int tid  = threadIdx.x;
    const int lane = tid & 31;
    const int wave = tid >> 5;          // nt = wave (0..3)
    const int lo16 = lane & 15;
    const int hi   = lane >> 4;

    const int ocp   = blockIdx.x * 64 + wave * 16 + lo16;  // packed column
    const int kbase = blockIdx.y * kchunk;
    float* pout = part + (size_t)blockIdx.y * ((size_t)B_DIM * OC_DIM);

    // A: per-lane K map K = j + 16*(lane>>4) -> two contiguous b128 runs
    const _Float16* aptr = xh + (size_t)lo16 * IC_DIM + kbase + hi * 16;
    // B: per-lane K map K = j + 16*(lane>>4) -> 16 contiguous ints (4x b128)
    const int* bptr = wp + (size_t)ocp * IC_DIM + kbase + hi * 16;

    v8f c[4][2];
#pragma unroll
    for (int mt = 0; mt < 4; ++mt) { c[mt][0] = (v8f)0.f; c[mt][1] = (v8f)0.f; }

    v4i  wbuf[2][4];     // raw packed weights, double buffered
    v16h abuf[2][4];     // A fragments (4 M tiles), double buffered

    // ---- prologue: fill buffer 0 (K offset 0) ----
#pragma unroll
    for (int g = 0; g < 4; ++g) wbuf[0][g] = *(const v4i*)(bptr + 4 * g);
#pragma unroll
    for (int mt = 0; mt < 4; ++mt) {
        const _Float16* ap = aptr + (size_t)mt * 16 * IC_DIM;
        v8h r0 = *(const v8h*)(ap);
        v8h r1 = *(const v8h*)(ap + 8);
        abuf[0][mt] = __builtin_shufflevector(r0, r1, 0, 1, 2, 3, 4, 5, 6, 7,
                                                      8, 9, 10, 11, 12, 13, 14, 15);
    }

    for (int k = 0; k < kchunk; k += 64) {
#pragma unroll
        for (int ph = 0; ph < 2; ++ph) {
            const int cur = ph, nxt = ph ^ 1;
            // Next tile's K offset: phase 0 -> +32 (always in range since
            // k+64 <= kchunk); phase 1 -> +64, clamped at the chunk boundary
            // (re-reads current tile; result unused after the loop).
            const int noff = (ph == 0) ? 32 : ((k + 64 < kchunk) ? 64 : 32);
            const int*      bnp = bptr + noff;
            const _Float16* anp = aptr + noff;

            // ---- issue NEXT tile's loads (in-flight across this phase) ----
#pragma unroll
            for (int g = 0; g < 4; ++g) wbuf[nxt][g] = *(const v4i*)(bnp + 4 * g);
#pragma unroll
            for (int mt = 0; mt < 4; ++mt) {
                const _Float16* ap = anp + (size_t)mt * 16 * IC_DIM;
                v8h r0 = *(const v8h*)(ap);
                v8h r1 = *(const v8h*)(ap + 8);
                abuf[nxt][mt] = __builtin_shufflevector(r0, r1,
                                    0, 1, 2, 3, 4, 5, 6, 7,
                                    8, 9, 10, 11, 12, 13, 14, 15);
            }
            __builtin_prefetch(bptr + noff + 256, 0, 1);   // HBM stream ahead

            // ---- convert CURRENT weights: 0x6400|q magic + pk_add(-1024) ----
            v8u plo, phi;
#pragma unroll
            for (int g = 0; g < 4; ++g) {
#pragma unroll
                for (int p = 0; p < 2; ++p) {
                    const int e0 = wbuf[cur][g][2 * p];
                    const int e1 = wbuf[cur][g][2 * p + 1];
                    const int i = g * 2 + p;
                    plo[i] = 0x64006400u | (unsigned)(e0 & 0xF)
                                         | ((unsigned)(e1 & 0xF) << 16);
                    phi[i] = 0x64006400u | (unsigned)((e0 >> 4) & 0xF)
                                         | ((unsigned)((e1 >> 4) & 0xF) << 16);
                }
            }
            v16h blo = __builtin_bit_cast(v16h, plo) + (_Float16)(-1024.0f);
            v16h bhi = __builtin_bit_cast(v16h, phi) + (_Float16)(-1024.0f);

            // ---- 8 WMMAs on CURRENT tile ----
#pragma unroll
            for (int mt = 0; mt < 4; ++mt) {
                c[mt][0] = __builtin_amdgcn_wmma_f32_16x16x32_f16(
                    false, abuf[cur][mt], false, blo, (short)0, c[mt][0], false, false);
                c[mt][1] = __builtin_amdgcn_wmma_f32_16x16x32_f16(
                    false, abuf[cur][mt], false, bhi, (short)0, c[mt][1], false, false);
            }
        }
        aptr += 64;
        bptr += 64;
    }

    // ---- write partial S (deterministic, disjoint regions per block) ----
#pragma unroll
    for (int mt = 0; mt < 4; ++mt) {
#pragma unroll
        for (int r = 0; r < 8; ++r) {
            const int m = mt * 16 + hi * 8 + r;    // C/D: M = r + 8*(lane>>4)
            pout[(size_t)m * OC_DIM + ocp]             = c[mt][0][r];
            pout[(size_t)m * OC_DIM + ocp + PACK_ROWS] = c[mt][1][r];
        }
    }
}

// ---------------------------------------------------------------------------
// Epilogue: out = delta*(S - zp*rowsum) + bias, fixed-order partial reduction.
// ---------------------------------------------------------------------------
__global__ __launch_bounds__(256) void w4_epilogue(
    const float* __restrict__ part, const float* __restrict__ rowsum,
    const float* __restrict__ delta, const float* __restrict__ zp,
    const float* __restrict__ bias, float* __restrict__ out, int ksplit)
{
    const int idx = blockIdx.x * 256 + threadIdx.x;   // 64*8192 elements
    const int m  = idx >> 13;
    const int oc = idx & (OC_DIM - 1);
    float s = 0.f;
    for (int p = 0; p < ksplit; ++p)
        s += part[(size_t)p * ((size_t)B_DIM * OC_DIM) + idx];
    out[idx] = delta[oc] * (s - zp[oc] * rowsum[m]) + bias[oc];
}

extern "C" void kernel_launch(void* const* d_in, const int* in_sizes, int n_in,
                              void* d_out, int out_size, void* d_ws, size_t ws_size,
                              hipStream_t stream) {
    const float* x     = (const float*)d_in[0];
    const int*   wp    = (const int*)d_in[1];
    const float* delta = (const float*)d_in[2];
    const float* zp    = (const float*)d_in[3];
    const float* bias  = (const float*)d_in[4];
    float* out = (float*)d_out;

    _Float16* xh     = (_Float16*)((char*)d_ws + XH_OFF);
    float*    rowsum = (float*)((char*)d_ws + RS_OFF);
    float*    part   = (float*)((char*)d_ws + PART_OFF);

    // K-split for occupancy; fixed for a given ws_size -> deterministic.
    int ksplit = 1;
    if (ws_size >= (size_t)PART_OFF + 4 * PART_SZ)      ksplit = 4;
    else if (ws_size >= (size_t)PART_OFF + 2 * PART_SZ) ksplit = 2;

    prep_x<<<B_DIM, 256, 0, stream>>>(x, xh, rowsum);

    dim3 grid(PACK_ROWS / 64, ksplit);                  // 64 col-blocks x ksplit
    w4_wmma_main<<<grid, 128, 0, stream>>>(xh, wp, part, IC_DIM / ksplit);

    w4_epilogue<<<(B_DIM * OC_DIM) / 256, 256, 0, stream>>>(
        part, rowsum, delta, zp, bias, out, ksplit);
}
